// PointTransformer_69672959475781
// MI455X (gfx1250) — matrix operand order repaired and checked by reading, not compiled, not run on hardware
//
#include <hip/hip_runtime.h>
#include <hip/hip_bf16.h>

// ---------------------------------------------------------------------------
// Point Transformer forward for gfx1250 (MI455X).
// Dense layers: LDS-staged bf16 tiles + v_wmma_f32_16x16x32_bf16 (fp32 acc).
// Workspace requirement: ~160 MB.
// ---------------------------------------------------------------------------

typedef __attribute__((ext_vector_type(16))) __bf16 v16bf;
typedef __attribute__((ext_vector_type(8)))  float  v8f;

#define LN_EPS 1e-5f

// LDS tile row stride in bf16 elements (80 bytes: 16B aligned, bank-staggered)
#define TSTRIDE 40

// ------------------------- WMMA fragment loaders (LDS) ---------------------
// A frag: 16x32 bf16 (MxK). Lane l: row m = l%16.
//   l<16 : VGPR0-3 hold K=0..7, VGPR4-7 hold K=16..23
//   l>=16: VGPR0-3 hold K=8..15, VGPR4-7 hold K=24..31
// -> two contiguous 8-element (16B) runs per lane: [half*8, +8) and [16+half*8, +8)
static __device__ __forceinline__ v16bf lds_frag_a(const __bf16* As, int mbase) {
  const int lane = threadIdx.x & 31;
  const __bf16* rowp = As + (mbase + (lane & 15)) * TSTRIDE;
  const int half = (lane >> 4) << 3;   // 0 or 8
  union { uint4 u[2]; v16bf v; } t;
  t.u[0] = *(const uint4*)(rowp + half);
  t.u[1] = *(const uint4*)(rowp + 16 + half);
  return t.v;
}

// B frag: 32x16 bf16 (KxN), stored transposed in LDS as [n][k].
//   l<16: K=0..7,16..23; l>=16: K=8..15,24..31 -> two 16B runs.
static __device__ __forceinline__ v16bf lds_frag_b(const __bf16* Bs, int nbase) {
  const int lane = threadIdx.x & 31;
  const __bf16* rowp = Bs + (nbase + (lane & 15)) * TSTRIDE + ((lane >> 4) << 3);
  union { uint4 u[2]; v16bf v; } t;
  t.u[0] = *(const uint4*)(rowp);
  t.u[1] = *(const uint4*)(rowp + 16);
  return t.v;
}

// C/D: 16x16 f32.  VGPR v: lanes 0-15 -> M=v, lanes 16-31 -> M=8+v.
static __device__ __forceinline__ void wmma_store_c(float* __restrict__ C, const v8f c,
                                                    int row, int col, int N,
                                                    const float* __restrict__ bias,
                                                    int relu) {
  const int lane  = threadIdx.x & 31;
  const int n     = col + (lane & 15);
  const int mbase = row + ((lane >> 4) << 3);
  const float bv  = bias ? bias[n] : 0.0f;
#pragma unroll
  for (int v = 0; v < 8; ++v) {
    float x = c[v] + bv;
    if (relu) x = fmaxf(x, 0.0f);
    C[(long long)(mbase + v) * N + n] = x;
  }
}

// ------------------------------ GEMM kernel --------------------------------
// C[M,N] = act(A[M,K] @ W[K,N] + bias).  M%64==0, N%64==0, K arbitrary,
// lda = physical row stride of A (lda%4==0 required whenever K>=32).
// 128 threads = 4 waves; block tile 64x64; each wave owns 32x32 (2x2 WMMA).
// Full K-tiles: float4 global loads (split load/store phases), b64 LDS stores
// for A. Tail tile: clamped scalar loads + select (no divergent loads).
__global__ __launch_bounds__(128)
void k_gemm(const float* __restrict__ A, const float* __restrict__ W,
            const float* __restrict__ bias, float* __restrict__ C,
            int M, int K, int lda, int N, int relu) {
  __shared__ __align__(16) __bf16 As[64 * TSTRIDE];
  __shared__ __align__(16) __bf16 Bs[64 * TSTRIDE];

  const int tid  = threadIdx.x;
  const int wave = tid >> 5;
  const int rowB = blockIdx.x * 64;
  const int colB = blockIdx.y * 64;
  const int wm = (wave >> 1) * 32;   // wave row offset within block tile
  const int wn = (wave & 1) * 32;    // wave col offset within block tile

  v8f c00 = {}, c01 = {}, c10 = {}, c11 = {};
  const int Kfull = K & ~31;

  for (int kb = 0; kb < K; kb += 32) {
    __syncthreads();
    if (kb < Kfull) {
      // --- full tile: float4 loads, phase-split so loads batch up ---
      float4 av[4], bv4[4];
#pragma unroll
      for (int i = 0; i < 4; ++i) {
        int e = (tid + i * 128) * 4;           // A: 64x32, element idx (mult 4)
        int mr = e >> 5, kk = e & 31;
        av[i] = *(const float4*)(A + (long long)(rowB + mr) * lda + kb + kk);
      }
#pragma unroll
      for (int i = 0; i < 4; ++i) {
        int e = (tid + i * 128) * 4;           // B: 32x64
        int kr = e >> 6, nn = e & 63;
        bv4[i] = *(const float4*)(W + (long long)(kb + kr) * N + colB + nn);
      }
      if (kb + 32 < K) {                        // pull next tiles toward WGP
        __builtin_prefetch(A + (long long)(rowB + (tid >> 2)) * lda + kb + 32, 0, 1);
        __builtin_prefetch(W + (long long)(kb + 32 + (tid >> 6)) * N + colB + (tid & 63), 0, 1);
      }
#pragma unroll
      for (int i = 0; i < 4; ++i) {
        int e = (tid + i * 128) * 4;
        int mr = e >> 5, kk = e & 31;
        __bf16 p[4] = {(__bf16)av[i].x, (__bf16)av[i].y, (__bf16)av[i].z, (__bf16)av[i].w};
        *(uint2*)(&As[mr * TSTRIDE + kk]) = *(const uint2*)p;   // ds_store_b64
      }
#pragma unroll
      for (int i = 0; i < 4; ++i) {
        int e = (tid + i * 128) * 4;
        int kr = e >> 6, nn = e & 63;
        Bs[(nn + 0) * TSTRIDE + kr] = (__bf16)bv4[i].x;          // transpose store
        Bs[(nn + 1) * TSTRIDE + kr] = (__bf16)bv4[i].y;
        Bs[(nn + 2) * TSTRIDE + kr] = (__bf16)bv4[i].z;
        Bs[(nn + 3) * TSTRIDE + kr] = (__bf16)bv4[i].w;
      }
    } else {
      // --- ragged tail: clamp address, select value, phase-split ---
      float ta[16], tb[16];
#pragma unroll
      for (int i = 0; i < 16; ++i) {
        int e = tid + i * 128;
        int mr = e >> 5, kk = e & 31;
        int kg = kb + kk;
        int kc = kg < K ? kg : K - 1;
        ta[i] = A[(long long)(rowB + mr) * lda + kc];
      }
#pragma unroll
      for (int i = 0; i < 16; ++i) {
        int e = tid + i * 128;
        int kr = e >> 6, nn = e & 63;
        int kg = kb + kr;
        int kc = kg < K ? kg : K - 1;
        tb[i] = W[(long long)kc * N + colB + nn];
      }
#pragma unroll
      for (int i = 0; i < 16; ++i) {
        int e = tid + i * 128;
        int mr = e >> 5, kk = e & 31;
        As[mr * TSTRIDE + kk] = (__bf16)((kb + kk) < K ? ta[i] : 0.0f);
      }
#pragma unroll
      for (int i = 0; i < 16; ++i) {
        int e = tid + i * 128;
        int kr = e >> 6, nn = e & 63;
        Bs[nn * TSTRIDE + kr] = (__bf16)((kb + kr) < K ? tb[i] : 0.0f);
      }
    }
    __syncthreads();

    v16bf a0 = lds_frag_a(As, wm);
    v16bf a1 = lds_frag_a(As, wm + 16);
    v16bf b0 = lds_frag_b(Bs, wn);
    v16bf b1 = lds_frag_b(Bs, wn + 16);
    c00 = __builtin_amdgcn_wmma_f32_16x16x32_bf16(false, a0, false, b0, (short)0, c00, false, false);
    c01 = __builtin_amdgcn_wmma_f32_16x16x32_bf16(false, a0, false, b1, (short)0, c01, false, false);
    c10 = __builtin_amdgcn_wmma_f32_16x16x32_bf16(false, a1, false, b0, (short)0, c10, false, false);
    c11 = __builtin_amdgcn_wmma_f32_16x16x32_bf16(false, a1, false, b1, (short)0, c11, false, false);
  }

  wmma_store_c(C, c00, rowB + wm,      colB + wn,      N, bias, relu);
  wmma_store_c(C, c01, rowB + wm,      colB + wn + 16, N, bias, relu);
  wmma_store_c(C, c10, rowB + wm + 16, colB + wn,      N, bias, relu);
  wmma_store_c(C, c11, rowB + wm + 16, colB + wn + 16, N, bias, relu);
}

// ------------------------------ LayerNorm ----------------------------------
// One block per row, blockDim == d (64/128/256, power of two).
__global__ void k_ln(const float* __restrict__ in, const float* __restrict__ g,
                     const float* __restrict__ b, float* __restrict__ out,
                     int d, int relu) {
  __shared__ float red[256];
  const long long row = blockIdx.x;
  const int c = threadIdx.x;
  float x = in[row * d + c];
  red[c] = x; __syncthreads();
  for (int s = d >> 1; s > 0; s >>= 1) { if (c < s) red[c] += red[c + s]; __syncthreads(); }
  float mu = red[0] / d; __syncthreads();
  float xm = x - mu;
  red[c] = xm * xm; __syncthreads();
  for (int s = d >> 1; s > 0; s >>= 1) { if (c < s) red[c] += red[c + s]; __syncthreads(); }
  float var = red[0] / d;
  float y = g[c] * xm * rsqrtf(var + LN_EPS) + b[c];
  if (relu) y = fmaxf(y, 0.0f);
  out[row * d + c] = y;
}

// -------------------------------- KNN --------------------------------------
// Brute force top-k (k<=16), LDS-tiled references. One thread per query.
#define KNN_TILE 256
__global__ void k_knn(const float* __restrict__ qry, const float* __restrict__ ref,
                      int Nq, int Nr, int kk, int* __restrict__ oidx,
                      float* __restrict__ odist) {
  __shared__ float sref[KNN_TILE * 3];
  const int b = blockIdx.y;
  const int q = blockIdx.x * blockDim.x + threadIdx.x;
  const float* qp = qry + (long long)b * Nq * 3;
  const float* rp = ref + (long long)b * Nr * 3;
  const bool act = q < Nq;
  float qx = 0, qy = 0, qz = 0;
  if (act) { qx = qp[q * 3]; qy = qp[q * 3 + 1]; qz = qp[q * 3 + 2]; }
  float bd[16]; int bi[16];
  for (int i = 0; i < 16; ++i) { bd[i] = 1e30f; bi[i] = 0; }
  for (int t0 = 0; t0 < Nr; t0 += KNN_TILE) {
    int tn = min(KNN_TILE, Nr - t0);
    __syncthreads();
    for (int i = threadIdx.x; i < tn * 3; i += blockDim.x) sref[i] = rp[t0 * 3 + i];
    __syncthreads();
    if (act) {
      for (int r = 0; r < tn; ++r) {
        float dx = qx - sref[r * 3], dy = qy - sref[r * 3 + 1], dz = qz - sref[r * 3 + 2];
        float d2 = dx * dx + dy * dy + dz * dz;
        if (d2 < bd[kk - 1]) {                      // strict < : earlier index wins ties
          int p = kk - 1;
          while (p > 0 && d2 < bd[p - 1]) { bd[p] = bd[p - 1]; bi[p] = bi[p - 1]; --p; }
          bd[p] = d2; bi[p] = t0 + r;
        }
      }
    }
  }
  if (act) {
    for (int i = 0; i < kk; ++i) {
      long long o = ((long long)b * Nq + q) * kk + i;
      oidx[o] = bi[i];
      if (odist) odist[o] = sqrtf(fmaxf(bd[i], 1e-12f));
    }
  }
}

// -------------------------------- FPS --------------------------------------
// One block per batch; serial npoint iterations, blocked point ownership so
// the strict-greater argmax reduction matches jnp.argmax (first max wins).
__global__ __launch_bounds__(512)
void k_fps(const float* __restrict__ xyz, int N, int npoint, int* __restrict__ out) {
  __shared__ float sval[512];
  __shared__ int   sidx[512];
  __shared__ float cent[3];
  const int b = blockIdx.x, t = threadIdx.x;
  const float* xp = xyz + (long long)b * N * 3;
  const int ppt = N / 512;
  float dist[8];
  for (int i = 0; i < 8; ++i) dist[i] = 1e10f;
  int far = 0;
  for (int it = 0; it < npoint; ++it) {
    if (t == 0) {
      out[(long long)b * npoint + it] = far;
      cent[0] = xp[far * 3]; cent[1] = xp[far * 3 + 1]; cent[2] = xp[far * 3 + 2];
    }
    __syncthreads();
    float bv = -1.0f; int bix = 0;
    for (int i = 0; i < ppt; ++i) {
      int p = t * ppt + i;
      float dx = xp[p * 3] - cent[0], dy = xp[p * 3 + 1] - cent[1], dz = xp[p * 3 + 2] - cent[2];
      float d = dx * dx + dy * dy + dz * dz;
      if (d < dist[i]) dist[i] = d;
      if (dist[i] > bv) { bv = dist[i]; bix = p; }
    }
    sval[t] = bv; sidx[t] = bix;
    __syncthreads();
    for (int s = 256; s > 0; s >>= 1) {
      if (t < s && sval[t + s] > sval[t]) { sval[t] = sval[t + s]; sidx[t] = sidx[t + s]; }
      __syncthreads();
    }
    far = sidx[0];
    __syncthreads();
  }
}

// --------------------------- small utilities -------------------------------
__global__ void k_extract_xyz(const float* __restrict__ x, float* __restrict__ xyz, int total) {
  int i = blockIdx.x * blockDim.x + threadIdx.x;
  if (i >= total) return;
  xyz[(long long)i * 3 + 0] = x[(long long)i * 6 + 0];
  xyz[(long long)i * 3 + 1] = x[(long long)i * 6 + 1];
  xyz[(long long)i * 3 + 2] = x[(long long)i * 6 + 2];
}

__global__ void k_gather_xyz(const float* __restrict__ xyz, const int* __restrict__ fidx,
                             float* __restrict__ out, int Nin, int Nout, int total) {
  int i = blockIdx.x * blockDim.x + threadIdx.x;
  if (i >= total) return;
  int b = i / Nout;
  const float* s = xyz + ((long long)b * Nin + fidx[i]) * 3;
  out[(long long)i * 3 + 0] = s[0];
  out[(long long)i * 3 + 1] = s[1];
  out[(long long)i * 3 + 2] = s[2];
}

// rel[r] = xyz[pt] - xyz[nbr], r = (pt-pt0)*16 + j.  Row stride 4 (16B rows).
__global__ void k_prep_rel(const float* __restrict__ xyz, const int* __restrict__ idx,
                           float* __restrict__ rel, int pt0, int rows, int Npts) {
  int r = blockIdx.x * blockDim.x + threadIdx.x;
  if (r >= rows) return;
  int pt = pt0 + (r >> 4), j = r & 15;
  int b = pt / Npts, n = pt % Npts;
  int nbr = idx[(long long)pt * 16 + j];
  const float* pc = xyz + ((long long)b * Npts + n) * 3;
  const float* pn = xyz + ((long long)b * Npts + nbr) * 3;
  rel[(long long)r * 4 + 0] = pc[0] - pn[0];
  rel[(long long)r * 4 + 1] = pc[1] - pn[1];
  rel[(long long)r * 4 + 2] = pc[2] - pn[2];
  rel[(long long)r * 4 + 3] = 0.0f;
}

// am_in = q[pt] - k[nbr] + pe
__global__ void k_am_in(const float* __restrict__ Q, const float* __restrict__ Km,
                        const float* __restrict__ pe, const int* __restrict__ idx,
                        float* __restrict__ out, int pt0, int rows, int Npts, int d) {
  long long e = (long long)blockIdx.x * blockDim.x + threadIdx.x;
  if (e >= (long long)rows * d) return;
  int r = (int)(e / d), c = (int)(e % d);
  int pt = pt0 + (r >> 4), j = r & 15;
  int b = pt / Npts;
  int nbr = idx[(long long)pt * 16 + j];
  out[e] = Q[(long long)pt * d + c] - Km[((long long)b * Npts + nbr) * d + c] + pe[e];
}

// per-channel softmax over 16 neighbors, weighted sum of (v+pe), residual, LN
__global__ void k_attn_finish(const float* __restrict__ logits, const float* __restrict__ pe,
                              const float* __restrict__ V, const int* __restrict__ idx,
                              const float* __restrict__ feat, const float* __restrict__ g,
                              const float* __restrict__ bta, float* __restrict__ out,
                              int pt0, int Npts, int d) {
  __shared__ float red[256];
  const int lp = blockIdx.x, c = threadIdx.x;
  const int pt = pt0 + lp;
  const int b = pt / Npts;
  float l[16];
  float m = -1e30f;
#pragma unroll
  for (int j = 0; j < 16; ++j) {
    l[j] = logits[((long long)lp * 16 + j) * d + c];
    m = fmaxf(m, l[j]);
  }
  float s = 0.0f;
#pragma unroll
  for (int j = 0; j < 16; ++j) { l[j] = __expf(l[j] - m); s += l[j]; }
  float inv = 1.0f / s;
  float acc = 0.0f;
#pragma unroll
  for (int j = 0; j < 16; ++j) {
    int nbr = idx[(long long)pt * 16 + j];
    float vv = V[((long long)b * Npts + nbr) * d + c] + pe[((long long)lp * 16 + j) * d + c];
    acc += vv * (l[j] * inv);
  }
  float x = acc + feat[(long long)pt * d + c];
  red[c] = x; __syncthreads();
  for (int s2 = d >> 1; s2 > 0; s2 >>= 1) { if (c < s2) red[c] += red[c + s2]; __syncthreads(); }
  float mu = red[0] / d; __syncthreads();
  float xm = x - mu;
  red[c] = xm * xm; __syncthreads();
  for (int s2 = d >> 1; s2 > 0; s2 >>= 1) { if (c < s2) red[c] += red[c + s2]; __syncthreads(); }
  float var = red[0] / d;
  out[(long long)pt * d + c] = g[c] * xm * rsqrtf(var + LN_EPS) + bta[c];
}

// concat(grp_xyz - new_xyz, grp_feat); output row stride dcp (padded, 16B rows)
__global__ void k_td_concat(const float* __restrict__ xyz, const float* __restrict__ newxyz,
                            const float* __restrict__ feat, const int* __restrict__ idx,
                            float* __restrict__ out, int p0, int rows, int Nref, int Nnew,
                            int d, int dc, int dcp) {
  long long e = (long long)blockIdx.x * blockDim.x + threadIdx.x;
  if (e >= (long long)rows * dcp) return;
  int r = (int)(e / dcp), c = (int)(e % dcp);
  int pnew = p0 + (r >> 4), j = r & 15;
  int b = pnew / Nnew;
  int nbr = idx[(long long)pnew * 16 + j];
  float val;
  if (c < 3)       val = xyz[((long long)b * Nref + nbr) * 3 + c] - newxyz[(long long)pnew * 3 + c];
  else if (c < dc) val = feat[((long long)b * Nref + nbr) * d + (c - 3)];
  else             val = 0.0f;
  out[e] = val;
}

// max over 16 group members, then LN
__global__ void k_maxpool_ln(const float* __restrict__ h, const float* __restrict__ g,
                             const float* __restrict__ bta, float* __restrict__ out,
                             int p0, int d) {
  __shared__ float red[256];
  const int lp = blockIdx.x, c = threadIdx.x;
  float x = -1e30f;
#pragma unroll
  for (int j = 0; j < 16; ++j) x = fmaxf(x, h[((long long)lp * 16 + j) * d + c]);
  red[c] = x; __syncthreads();
  for (int s = d >> 1; s > 0; s >>= 1) { if (c < s) red[c] += red[c + s]; __syncthreads(); }
  float mu = red[0] / d; __syncthreads();
  float xm = x - mu;
  red[c] = xm * xm; __syncthreads();
  for (int s = d >> 1; s > 0; s >>= 1) { if (c < s) red[c] += red[c + s]; __syncthreads(); }
  float var = red[0] / d;
  out[(long long)(p0 + lp) * d + c] = g[c] * xm * rsqrtf(var + LN_EPS) + bta[c];
}

// concat(skip, inverse-distance 3-NN interpolation of feat_lo)
__global__ void k_interp_concat(const float* __restrict__ skip, const float* __restrict__ flo,
                                const int* __restrict__ idx3, const float* __restrict__ dist3,
                                float* __restrict__ out, int Nhi, int Nlo, int ds, int dl,
                                int total) {
  const int dc = ds + dl;
  long long e = (long long)blockIdx.x * blockDim.x + threadIdx.x;
  if (e >= (long long)total * dc) return;
  int r = (int)(e / dc), c = (int)(e % dc);
  if (c < ds) { out[e] = skip[(long long)r * ds + c]; return; }
  int b = r / Nhi;
  float i0 = 1.0f / (dist3[(long long)r * 3 + 0] + 1e-8f);
  float i1 = 1.0f / (dist3[(long long)r * 3 + 1] + 1e-8f);
  float i2 = 1.0f / (dist3[(long long)r * 3 + 2] + 1e-8f);
  float wsum = i0 + i1 + i2;
  int cc = c - ds;
  float v = (i0 * flo[((long long)b * Nlo + idx3[(long long)r * 3 + 0]) * dl + cc] +
             i1 * flo[((long long)b * Nlo + idx3[(long long)r * 3 + 1]) * dl + cc] +
             i2 * flo[((long long)b * Nlo + idx3[(long long)r * 3 + 2]) * dl + cc]) / wsum;
  out[e] = v;
}

// --------------------------------------------------------------------------
//                                host side
// --------------------------------------------------------------------------
struct PtP {
  const float *am0_b, *am0_w, *am1_b, *am1_w, *ln_b, *ln_g;
  const float *pe0_b, *pe0_w, *pe1_b, *pe1_w, *wk, *wq, *wv;
};

extern "C" void kernel_launch(void* const* d_in, const int* in_sizes, int n_in,
                              void* d_out, int out_size, void* d_ws, size_t ws_size,
                              hipStream_t stream) {
  const int Bb = 8, N0 = 4096, N1 = 1024, N2 = 256;
  const int D0 = 64, D1 = 128, D2 = 256;
  const int TOT0 = Bb * N0, TOT1 = Bb * N1, TOT2 = Bb * N2;

  auto in_f = [&](int i) { return (const float*)d_in[i]; };

  // Locate x by its unique flat size; JAX tree-flatten sorts dict keys, so
  // normally params (97 leaves) come first and x is last.
  int xi = n_in - 1;
  for (int i = 0; i < n_in; ++i)
    if (in_sizes[i] == TOT0 * 6) { xi = i; break; }
  const float* x = in_f(xi);
  const int pb = (xi == 0) ? 1 : 0;  // params base offset

  auto getpt = [&](int base) {
    PtP p;
    p.am0_b = in_f(base + 0);  p.am0_w = in_f(base + 1);
    p.am1_b = in_f(base + 2);  p.am1_w = in_f(base + 3);
    p.ln_b  = in_f(base + 4);  p.ln_g  = in_f(base + 5);
    p.pe0_b = in_f(base + 6);  p.pe0_w = in_f(base + 7);
    p.pe1_b = in_f(base + 8);  p.pe1_w = in_f(base + 9);
    p.wk = in_f(base + 10); p.wq = in_f(base + 11); p.wv = in_f(base + 12);
    return p;
  };
  PtP dec1 = getpt(pb + 0), dec2 = getpt(pb + 13);
  PtP enc0 = getpt(pb + 26), enc1 = getpt(pb + 39), enc2 = getpt(pb + 52);
  const float *head_b = in_f(pb + 65), *head_w = in_f(pb + 66);
  const float *head_lnb = in_f(pb + 67), *head_lng = in_f(pb + 68);
  const float *stem_b = in_f(pb + 69), *stem_w = in_f(pb + 70);
  const float *stem_lnb = in_f(pb + 71), *stem_lng = in_f(pb + 72);
  const float *td1_lnb = in_f(pb + 73), *td1_lng = in_f(pb + 74);
  const float *td1_m0b = in_f(pb + 75), *td1_m0w = in_f(pb + 76);
  const float *td1_m1b = in_f(pb + 77), *td1_m1w = in_f(pb + 78);
  const float *td2_lnb = in_f(pb + 79), *td2_lng = in_f(pb + 80);
  const float *td2_m0b = in_f(pb + 81), *td2_m0w = in_f(pb + 82);
  const float *td2_m1b = in_f(pb + 83), *td2_m1w = in_f(pb + 84);
  const float *tu1_f0b = in_f(pb + 85), *tu1_f0w = in_f(pb + 86);
  const float *tu1_f1b = in_f(pb + 87), *tu1_f1w = in_f(pb + 88);
  const float *tu1_lnb = in_f(pb + 89), *tu1_lng = in_f(pb + 90);
  const float *tu2_f0b = in_f(pb + 91), *tu2_f0w = in_f(pb + 92);
  const float *tu2_f1b = in_f(pb + 93), *tu2_f1w = in_f(pb + 94);
  const float *tu2_lnb = in_f(pb + 95), *tu2_lng = in_f(pb + 96);

  // ------------------------ workspace carving ------------------------------
  char* ws = (char*)d_ws;
  size_t off = 0;
  auto carve = [&](size_t bytes) {
    void* p = ws + off;
    off = (off + bytes + 255) & ~(size_t)255;
    return p;
  };
  float* XYZ0 = (float*)carve((size_t)TOT0 * 3 * 4);
  float* XYZ1 = (float*)carve((size_t)TOT1 * 3 * 4);
  float* XYZ2 = (float*)carve((size_t)TOT2 * 3 * 4);
  float* F0A = (float*)carve((size_t)TOT0 * D0 * 4);
  float* F0B = (float*)carve((size_t)TOT0 * D0 * 4);
  float* F1A = (float*)carve((size_t)TOT1 * D1 * 4);
  float* F1B = (float*)carve((size_t)TOT1 * D1 * 4);
  float* F2A = (float*)carve((size_t)TOT2 * D2 * 4);
  float* F2B = (float*)carve((size_t)TOT2 * D2 * 4);
  float* QB = (float*)carve((size_t)TOT0 * D0 * 4);
  float* KB = (float*)carve((size_t)TOT0 * D0 * 4);
  float* VB = (float*)carve((size_t)TOT0 * D0 * 4);
  int*   IDX16 = (int*)carve((size_t)TOT0 * 16 * 4);
  int*   IDX3  = (int*)carve((size_t)TOT0 * 3 * 4);
  float* DIST3 = (float*)carve((size_t)TOT0 * 3 * 4);
  int*   FIDX  = (int*)carve((size_t)TOT1 * 4);
  float* REL   = (float*)carve((size_t)65536 * 4 * 4);
  const size_t BUFSZ = (size_t)34 * 1024 * 1024;
  float* BUF0 = (float*)carve(BUFSZ);
  float* BUF1 = (float*)carve(BUFSZ);
  float* BUF2 = (float*)carve(BUFSZ);
  (void)ws_size; (void)out_size; (void)n_in;

  auto gemm = [&](const float* A, const float* W, const float* bias, float* C,
                  int M, int K, int lda, int Nc, int relu) {
    dim3 g(M / 64, Nc / 64);
    k_gemm<<<g, 128, 0, stream>>>(A, W, bias, C, M, K, lda, Nc, relu);
  };

  auto pt_layer = [&](const PtP& p, const float* xyz, const float* fin, float* fout,
                      int Npts, int d) {
    const int tot = Bb * Npts;
    gemm(fin, p.wq, nullptr, QB, tot, d, d, d, 0);
    gemm(fin, p.wk, nullptr, KB, tot, d, d, d, 0);
    gemm(fin, p.wv, nullptr, VB, tot, d, d, d, 0);
    { dim3 g((Npts + 255) / 256, Bb);
      k_knn<<<g, 256, 0, stream>>>(xyz, xyz, Npts, Npts, 16, IDX16, nullptr); }
    const int chunk = (tot > 4096) ? 4096 : tot;
    for (int p0 = 0; p0 < tot; p0 += chunk) {
      const int rows = chunk * 16;
      k_prep_rel<<<(rows + 255) / 256, 256, 0, stream>>>(xyz, IDX16, REL, p0, rows, Npts);
      gemm(REL, p.pe0_w, p.pe0_b, BUF0, rows, 3, 4, d, 1);     // pe hidden (relu)
      gemm(BUF0, p.pe1_w, p.pe1_b, BUF1, rows, d, d, d, 0);    // pe
      { long long te = (long long)rows * d;
        k_am_in<<<(unsigned)((te + 255) / 256), 256, 0, stream>>>(
            QB, KB, BUF1, IDX16, BUF0, p0, rows, Npts, d); }
      gemm(BUF0, p.am0_w, p.am0_b, BUF2, rows, d, d, d, 1);    // am hidden (relu)
      gemm(BUF2, p.am1_w, p.am1_b, BUF0, rows, d, d, d, 0);    // logits
      k_attn_finish<<<chunk, d, 0, stream>>>(BUF0, BUF1, VB, IDX16, fin,
                                             p.ln_g, p.ln_b, fout, p0, Npts, d);
    }
  };

  auto trans_down = [&](const float* xyz_in, int Nin, const float* fin, int din,
                        float* xyz_out, int Nout, float* fout, int dout,
                        const float* m0b, const float* m0w,
                        const float* m1b, const float* m1w,
                        const float* lng, const float* lnb) {
    k_fps<<<Bb, 512, 0, stream>>>(xyz_in, Nin, Nout, FIDX);
    const int tot = Bb * Nout;
    k_gather_xyz<<<(tot + 255) / 256, 256, 0, stream>>>(xyz_in, FIDX, xyz_out, Nin, Nout, tot);
    { dim3 g((Nout + 255) / 256, Bb);
      k_knn<<<g, 256, 0, stream>>>(xyz_out, xyz_in, Nout, Nin, 16, IDX16, nullptr); }
    const int dc = 3 + din;
    const int dcp = (dc + 3) & ~3;   // padded row stride (16B aligned rows)
    const int chunk = (tot > 4096) ? 4096 : tot;
    for (int p0 = 0; p0 < tot; p0 += chunk) {
      const int rows = chunk * 16;
      long long te = (long long)rows * dcp;
      k_td_concat<<<(unsigned)((te + 255) / 256), 256, 0, stream>>>(
          xyz_in, xyz_out, fin, IDX16, BUF0, p0, rows, Nin, Nout, din, dc, dcp);
      gemm(BUF0, m0w, m0b, BUF1, rows, dc, dcp, dout, 1);
      gemm(BUF1, m1w, m1b, BUF2, rows, dout, dout, dout, 0);
      k_maxpool_ln<<<chunk, dout, 0, stream>>>(BUF2, lng, lnb, fout, p0, dout);
    }
  };

  auto trans_up = [&](const float* xyz_hi, int Nhi, const float* xyz_lo, int Nlo,
                      const float* fskip, int ds, const float* flo, int dl,
                      float* fout,
                      const float* f0b, const float* f0w,
                      const float* f1b, const float* f1w,
                      const float* lng, const float* lnb) {
    { dim3 g((Nhi + 255) / 256, Bb);
      k_knn<<<g, 256, 0, stream>>>(xyz_hi, xyz_lo, Nhi, Nlo, 3, IDX3, DIST3); }
    const int tot = Bb * Nhi;
    const int dc = ds + dl;          // 384 / 192, already multiples of 4
    long long te = (long long)tot * dc;
    k_interp_concat<<<(unsigned)((te + 255) / 256), 256, 0, stream>>>(
        fskip, flo, IDX3, DIST3, BUF0, Nhi, Nlo, ds, dl, tot);
    gemm(BUF0, f0w, f0b, BUF1, tot, dc, dc, ds, 1);
    gemm(BUF1, f1w, f1b, BUF2, tot, ds, ds, ds, 0);
    k_ln<<<tot, ds, 0, stream>>>(BUF2, lng, lnb, fout, ds, 0);
  };

  // ------------------------------- pipeline --------------------------------
  k_extract_xyz<<<(TOT0 + 255) / 256, 256, 0, stream>>>(x, XYZ0, TOT0);

  // stem: relu(LN(x @ W + b))  (K=6 < 32: tail path only, lda=6 is fine)
  gemm(x, stem_w, stem_b, BUF0, TOT0, 6, 6, D0, 0);
  k_ln<<<TOT0, D0, 0, stream>>>(BUF0, stem_lng, stem_lnb, F0A, D0, 1);

  pt_layer(enc0, XYZ0, F0A, F0B, N0, D0);                         // f0 -> F0B
  trans_down(XYZ0, N0, F0B, D0, XYZ1, N1, F1A, D1,
             td1_m0b, td1_m0w, td1_m1b, td1_m1w, td1_lng, td1_lnb);
  pt_layer(enc1, XYZ1, F1A, F1B, N1, D1);                         // f1 -> F1B
  trans_down(XYZ1, N1, F1B, D1, XYZ2, N2, F2A, D2,
             td2_m0b, td2_m0w, td2_m1b, td2_m1w, td2_lng, td2_lnb);
  pt_layer(enc2, XYZ2, F2A, F2B, N2, D2);                         // f2 -> F2B

  trans_up(XYZ1, N1, XYZ2, N2, F1B, D1, F2B, D2, F1A,
           tu1_f0b, tu1_f0w, tu1_f1b, tu1_f1w, tu1_lng, tu1_lnb); // f1d -> F1A
  pt_layer(dec1, XYZ1, F1A, F1B, N1, D1);                         // -> F1B
  trans_up(XYZ0, N0, XYZ1, N1, F0B, D0, F1B, D1, F0A,
           tu2_f0b, tu2_f0w, tu2_f1b, tu2_f1w, tu2_lng, tu2_lnb); // f0d -> F0A
  pt_layer(dec2, XYZ0, F0A, F0B, N0, D0);                         // -> F0B

  // head: LN(f0d @ W + b)
  gemm(F0B, head_w, head_b, BUF0, TOT0, D0, D0, D2, 0);
  k_ln<<<TOT0, D2, 0, stream>>>(BUF0, head_lng, head_lnb, (float*)d_out, D2, 0);
}
